// ResidualAttentionBlock_53661321396764
// MI455X (gfx1250) — matrix operand order repaired and checked by reading, not compiled, not run on hardware
//
#include <hip/hip_runtime.h>
#include <math.h>

// ---------------------------------------------------------------------------
// Types for CDNA5 WMMA (wave32): V_WMMA_F32_16X16X32_BF16
// ---------------------------------------------------------------------------
typedef __bf16 bf16;
typedef bf16  v16bf __attribute__((ext_vector_type(16)));
typedef bf16  v8bf  __attribute__((ext_vector_type(8)));
typedef float v8f   __attribute__((ext_vector_type(8)));

__device__ inline v16bf make_v16(v8bf lo, v8bf hi) {
    union { v16bf v; v8bf h[2]; } u;
    u.h[0] = lo; u.h[1] = hi;
    return u.v;
}

// ---------------------------------------------------------------------------
// Generic bf16 WMMA GEMM:  C = scale * (A @ B) + bias
//   A  : M x K bf16, row-major, leading dim lda, batch stride sA
//   Bt : N x K bf16, row-major (i.e. B transposed), ldb, batch stride sB
// OUT template:
//   0 : bf16 row-major output (ldc, batch stride sC)
//   1 : bf16 transposed output  out[b][n][s] with b = m/seq, s = m%seq
//       (tBatchStride = N*seq); grid.z must be 1
//   2 : fp32 row-major output (ldc, batch stride sC)
//
// Block: 128 threads = 4 waves; wave grid 2x2; wave tile 64x64; block 128x128.
// Fragment layouts per CDNA5 ISA 7.12.2 (16-bit A 16x32, B as Bt rows).
// ---------------------------------------------------------------------------
template <int OUT>
__global__ __launch_bounds__(128)
void gemm_wmma_bf16(const bf16* __restrict__ A,
                    const bf16* __restrict__ Bt,
                    const float* __restrict__ bias,
                    float scale,
                    void* __restrict__ Cout,
                    int K, int lda, int ldb, int ldc,
                    long long sA, long long sB, long long sC,
                    int seq, long long tBatchStride)
{
    const int z = blockIdx.z;
    A  += (size_t)z * sA;
    Bt += (size_t)z * sB;

    const int lane  = threadIdx.x & 31;
    const int wave  = threadIdx.x >> 5;
    const int waveM = wave >> 1;
    const int waveN = wave & 1;
    const int m0 = blockIdx.y * 128 + waveM * 64;
    const int n0 = blockIdx.x * 128 + waveN * 64;
    const int lh = lane >> 4;   // lane half: 0 or 1
    const int lm = lane & 15;

    v8f acc[4][4];
#pragma unroll
    for (int i = 0; i < 4; ++i)
#pragma unroll
        for (int j = 0; j < 4; ++j)
#pragma unroll
            for (int e = 0; e < 8; ++e)
                acc[i][j][e] = 0.0f;

    for (int k0 = 0; k0 < K; k0 += 32) {
        v16bf af[4], bf[4];
        // A fragment: lane half 0 holds K = {0..7, 16..23}, half 1 K = {8..15, 24..31}
#pragma unroll
        for (int mi = 0; mi < 4; ++mi) {
            const bf16* pa = A + (size_t)(m0 + mi * 16 + lm) * lda + k0 + lh * 8;
            v8bf lo = *(const v8bf*)pa;
            v8bf hi = *(const v8bf*)(pa + 16);
            af[mi] = make_v16(lo, hi);
            __builtin_prefetch(pa + 32, 0, 0);       // global_prefetch next K-step
        }
        // B fragment from Bt: lane half 0 holds K = {0..7, 16..23}, half 1 K = {8..15, 24..31}
#pragma unroll
        for (int ni = 0; ni < 4; ++ni) {
            const bf16* pb = Bt + (size_t)(n0 + ni * 16 + lm) * ldb + k0 + lh * 8;
            bf[ni] = make_v16(*(const v8bf*)pb, *(const v8bf*)(pb + 16));
            __builtin_prefetch(pb + 32, 0, 0);
        }
#pragma unroll
        for (int mi = 0; mi < 4; ++mi)
#pragma unroll
            for (int ni = 0; ni < 4; ++ni)
                acc[mi][ni] = __builtin_amdgcn_wmma_f32_16x16x32_bf16(
                    false, af[mi], false, bf[ni], (short)0, acc[mi][ni], false, false);
    }

    // Epilogue. C fragment: lane L, VGPR r -> (M = r + (L>=16)*8, N = L%16)
#pragma unroll
    for (int mi = 0; mi < 4; ++mi) {
#pragma unroll
        for (int ni = 0; ni < 4; ++ni) {
            const int n     = n0 + ni * 16 + lm;
            const int mbase = m0 + mi * 16 + lh * 8;
            const float bv  = bias ? bias[n] : 0.0f;
            if (OUT == 1) {
                v8bf pack;
#pragma unroll
                for (int r = 0; r < 8; ++r)
                    pack[r] = (bf16)(acc[mi][ni][r] * scale + bv);
                const int bidx = mbase / seq;
                const int s    = mbase % seq;
                bf16* ct = (bf16*)Cout + (size_t)bidx * tBatchStride
                                       + (size_t)n * seq + s;
                *(v8bf*)ct = pack;
            } else if (OUT == 0) {
                bf16* cb = (bf16*)Cout + (size_t)z * sC;
#pragma unroll
                for (int r = 0; r < 8; ++r)
                    cb[(size_t)(mbase + r) * ldc + n] =
                        (bf16)(acc[mi][ni][r] * scale + bv);
            } else {
                float* cf = (float*)Cout + (size_t)z * sC;
#pragma unroll
                for (int r = 0; r < 8; ++r)
                    cf[(size_t)(mbase + r) * ldc + n] =
                        acc[mi][ni][r] * scale + bv;
            }
        }
    }
}

// ---------------------------------------------------------------------------
// fp32 -> bf16 flat convert
// ---------------------------------------------------------------------------
__global__ __launch_bounds__(256)
void f32_to_bf16(const float* __restrict__ in, bf16* __restrict__ out, size_t n)
{
    size_t i = (size_t)blockIdx.x * blockDim.x + threadIdx.x;
    if (i < n) out[i] = (bf16)in[i];
}

// ---------------------------------------------------------------------------
// Transposed convert: W (R x C fp32, row-major) -> Wt (C x R bf16, row-major)
// ---------------------------------------------------------------------------
__global__ __launch_bounds__(256)
void transpose_to_bf16(const float* __restrict__ W, bf16* __restrict__ Wt,
                       int R, int C)
{
    int idx = blockIdx.x * blockDim.x + threadIdx.x;
    if (idx >= R * C) return;
    int n = idx / R;  // output row (original column)
    int k = idx % R;
    Wt[(size_t)n * R + k] = (bf16)W[(size_t)k * C + n];
}

// ---------------------------------------------------------------------------
// Row softmax: normalize fp32 scores in place (d_out attn region) and emit a
// bf16 copy for the attn @ V GEMM. One block per row, S <= 2048.
// ---------------------------------------------------------------------------
__global__ __launch_bounds__(256)
void softmax_rows(float* __restrict__ attn, bf16* __restrict__ attnb, int S)
{
    const size_t row = blockIdx.x;
    float* p  = attn  + row * (size_t)S;
    bf16*  pb = attnb + row * (size_t)S;
    const int t = threadIdx.x;

    __shared__ float red[256];

    float e[8];
    float mx = -INFINITY;
#pragma unroll
    for (int j = 0; j < 8; ++j) {
        int idx = t + j * 256;
        e[j] = (idx < S) ? p[idx] : -INFINITY;
        mx = fmaxf(mx, e[j]);
    }
    red[t] = mx; __syncthreads();
    for (int s = 128; s > 0; s >>= 1) {
        if (t < s) red[t] = fmaxf(red[t], red[t + s]);
        __syncthreads();
    }
    mx = red[0]; __syncthreads();

    float sum = 0.0f;
#pragma unroll
    for (int j = 0; j < 8; ++j) {
        int idx = t + j * 256;
        if (idx < S) { e[j] = __expf(e[j] - mx); sum += e[j]; }
    }
    red[t] = sum; __syncthreads();
    for (int s = 128; s > 0; s >>= 1) {
        if (t < s) red[t] += red[t + s];
        __syncthreads();
    }
    const float inv = 1.0f / red[0];

#pragma unroll
    for (int j = 0; j < 8; ++j) {
        int idx = t + j * 256;
        if (idx < S) {
            float a = e[j] * inv;
            p[idx]  = a;
            pb[idx] = (bf16)a;
        }
    }
}

// ---------------------------------------------------------------------------
// Residual add + LayerNorm over D=512. One block of 256 threads per row.
// ---------------------------------------------------------------------------
__global__ __launch_bounds__(256)
void resid_layernorm(const float* __restrict__ proj, const float* __restrict__ x,
                     const float* __restrict__ gamma, const float* __restrict__ beta,
                     float* __restrict__ y, int D)
{
    const size_t row = blockIdx.x;
    const float* pp = proj + row * (size_t)D;
    const float* px = x    + row * (size_t)D;
    float*       py = y    + row * (size_t)D;
    const int t = threadIdx.x;

    __shared__ float red[256];

    float v0 = pp[t]       + px[t];
    float v1 = pp[t + 256] + px[t + 256];

    red[t] = v0 + v1; __syncthreads();
    for (int s = 128; s > 0; s >>= 1) {
        if (t < s) red[t] += red[t + s];
        __syncthreads();
    }
    const float mean = red[0] / (float)D;
    __syncthreads();

    const float d0 = v0 - mean, d1 = v1 - mean;
    red[t] = d0 * d0 + d1 * d1; __syncthreads();
    for (int s = 128; s > 0; s >>= 1) {
        if (t < s) red[t] += red[t + s];
        __syncthreads();
    }
    const float rstd = rsqrtf(red[0] / (float)D + 1e-5f);

    py[t]       = d0 * rstd * gamma[t]       + beta[t];
    py[t + 256] = d1 * rstd * gamma[t + 256] + beta[t + 256];
}

// ---------------------------------------------------------------------------
// Host launcher
// ---------------------------------------------------------------------------
extern "C" void kernel_launch(void* const* d_in, const int* in_sizes, int n_in,
                              void* d_out, int out_size, void* d_ws, size_t ws_size,
                              hipStream_t stream)
{
    (void)in_sizes; (void)n_in; (void)out_size; (void)ws_size;

    const float* x     = (const float*)d_in[0];
    const float* Wq    = (const float*)d_in[1];
    const float* bq    = (const float*)d_in[2];
    const float* Wk    = (const float*)d_in[3];
    const float* bk    = (const float*)d_in[4];
    const float* Wv    = (const float*)d_in[5];
    const float* bv    = (const float*)d_in[6];
    const float* Wp    = (const float*)d_in[7];
    const float* bp    = (const float*)d_in[8];
    const float* gamma = (const float*)d_in[9];
    const float* beta  = (const float*)d_in[10];

    const int B = 8, S = 2048, D = 512, H = 512;
    const size_t M = (size_t)B * S;            // 16384

    // Output layout: y (M*D fp32) then attn (B*S*S fp32)
    float* y_out    = (float*)d_out;
    float* attn_out = (float*)d_out + M * D;

    // Workspace carve-up (bf16 = 2 bytes)
    char* ws = (char*)d_ws;
    size_t off = 0;
    auto carve = [&](size_t bytes) { char* p = ws + off; off += (bytes + 255) & ~(size_t)255; return p; };
    bf16*  x_b    = (bf16*)carve(M * D * 2);
    bf16*  q_b    = (bf16*)carve(M * H * 2);
    bf16*  k_b    = (bf16*)carve(M * H * 2);
    bf16*  vT_b   = (bf16*)carve((size_t)B * H * S * 2);   // [b][h][s]
    bf16*  attn_b = (bf16*)carve((size_t)B * S * S * 2);
    bf16*  o_b    = (bf16*)carve(M * H * 2);
    float* proj_f = (float*)carve(M * D * 4);
    bf16*  WqT    = (bf16*)carve((size_t)H * D * 2);
    bf16*  WkT    = (bf16*)carve((size_t)H * D * 2);
    bf16*  WvT    = (bf16*)carve((size_t)H * D * 2);
    bf16*  WpT    = (bf16*)carve((size_t)D * H * 2);

    const float inv_scale = 1.0f / sqrtf((float)H);

    // 1) Conversions
    {
        size_t n = M * D;
        f32_to_bf16<<<dim3((unsigned)((n + 255) / 256)), dim3(256), 0, stream>>>(x, x_b, n);
        int wn = D * H;
        dim3 wg((wn + 255) / 256), wb(256);
        transpose_to_bf16<<<wg, wb, 0, stream>>>(Wq, WqT, D, H);
        transpose_to_bf16<<<wg, wb, 0, stream>>>(Wk, WkT, D, H);
        transpose_to_bf16<<<wg, wb, 0, stream>>>(Wv, WvT, D, H);
        transpose_to_bf16<<<wg, wb, 0, stream>>>(Wp, WpT, H, D);
    }

    const dim3 blk(128);

    // 2) q = x @ Wq + bq   (bf16 out)
    gemm_wmma_bf16<0><<<dim3(H / 128, M / 128, 1), blk, 0, stream>>>(
        x_b, WqT, bq, 1.0f, (void*)q_b, D, D, D, H, 0, 0, 0, 0, 0);
    // 3) k = x @ Wk + bk   (bf16 out)
    gemm_wmma_bf16<0><<<dim3(H / 128, M / 128, 1), blk, 0, stream>>>(
        x_b, WkT, bk, 1.0f, (void*)k_b, D, D, D, H, 0, 0, 0, 0, 0);
    // 4) v = x @ Wv + bv   (bf16, stored transposed per batch: vT[b][h][s])
    gemm_wmma_bf16<1><<<dim3(H / 128, M / 128, 1), blk, 0, stream>>>(
        x_b, WvT, bv, 1.0f, (void*)vT_b, D, D, D, 0, 0, 0, 0,
        S, (long long)H * S);

    // 5) scores = q @ k^T / sqrt(H)  (fp32 into d_out attn region), batched
    gemm_wmma_bf16<2><<<dim3(S / 128, S / 128, B), blk, 0, stream>>>(
        q_b, k_b, nullptr, inv_scale, (void*)attn_out, H, H, H, S,
        (long long)S * H, (long long)S * H, (long long)S * S, 0, 0);

    // 6) softmax rows (in-place fp32 + bf16 copy)
    softmax_rows<<<dim3((unsigned)(B * S)), dim3(256), 0, stream>>>(attn_out, attn_b, S);

    // 7) o = attn @ v   (bf16 out), batched; Bt = vT[b]
    gemm_wmma_bf16<0><<<dim3(H / 128, S / 128, B), blk, 0, stream>>>(
        attn_b, vT_b, nullptr, 1.0f, (void*)o_b, S, S, S, H,
        (long long)S * S, (long long)H * S, (long long)S * H, 0, 0);

    // 8) proj = o @ Wp + bp   (fp32 out)
    gemm_wmma_bf16<2><<<dim3(D / 128, M / 128, 1), blk, 0, stream>>>(
        o_b, WpT, bp, 1.0f, (void*)proj_f, H, H, H, D, 0, 0, 0, 0, 0);

    // 9) y = LayerNorm(proj + x) * gamma + beta
    resid_layernorm<<<dim3((unsigned)M), dim3(256), 0, stream>>>(
        proj_f, x, gamma, beta, y_out, D);
}
